// TorchairDeepseekV2MoE_11596411699304
// MI455X (gfx1250) — matrix operand order repaired and checked by reading, not compiled, not run on hardware
//
#include <hip/hip_runtime.h>

#define T_TOK 1024
#define H_DIM 2048
#define E_EXP 64
#define I_DIM 1024
#define NGROUP 8
#define GSIZE 8            // E / NGROUP
#define TOPKG 4
#define TOPK 8
#define IS_DIM 2048        // I * N_SHARED
#define RSCALE 2.5f

typedef __attribute__((ext_vector_type(16))) _Float16 v16h;
typedef __attribute__((ext_vector_type(8)))  float    v8f;

// ---------------------------------------------------------------- helpers ---

__device__ __forceinline__ v8f wmma_f16(v16h a, v16h b, v8f c) {
  // 8 args: (neg_a, A, neg_b, B, c_mod, C, reuse_a, reuse_b)
  return __builtin_amdgcn_wmma_f32_16x16x32_f16(false, a, false, b, (short)0, c,
                                                false, false);
}

__device__ __forceinline__ v16h cvt16(float4 a, float4 b, float4 c, float4 d) {
  v16h v;
  v[0]  = (_Float16)a.x; v[1]  = (_Float16)a.y; v[2]  = (_Float16)a.z; v[3]  = (_Float16)a.w;
  v[4]  = (_Float16)b.x; v[5]  = (_Float16)b.y; v[6]  = (_Float16)b.z; v[7]  = (_Float16)b.w;
  v[8]  = (_Float16)c.x; v[9]  = (_Float16)c.y; v[10] = (_Float16)c.z; v[11] = (_Float16)c.w;
  v[12] = (_Float16)d.x; v[13] = (_Float16)d.y; v[14] = (_Float16)d.z; v[15] = (_Float16)d.w;
  return v;
}

// A fragment (16x32 f16): lane m = lane&15, hi = lane>>4.
// halves j=0..7 -> k = kbase + hi*8 + j ; j=8..15 -> k = kbase + 16 + hi*8 + (j-8)
__device__ __forceinline__ v16h load_a_f32(const float* __restrict__ row, int kbase, int hi) {
  const float* p0 = row + kbase + hi * 8;
  const float* p1 = row + kbase + 16 + hi * 8;
  return cvt16(*(const float4*)(p0), *(const float4*)(p0 + 4),
               *(const float4*)(p1), *(const float4*)(p1 + 4));
}

// B fragment (32x16 f16): lane n = lane&15, hi = lane>>4.
// halves j=0..7 -> k = kbase + hi*8 + j ; j=8..15 -> k = kbase + 16 + hi*8 + (j-8).  nrow = W + (nbase + (lane&15)) * ld
__device__ __forceinline__ v16h load_b_f32(const float* __restrict__ nrow, int kbase, int hi) {
  const float* p0 = nrow + kbase + hi * 8;
  const float* p1 = nrow + kbase + 16 + hi * 8;
  return cvt16(*(const float4*)(p0), *(const float4*)(p0 + 4),
               *(const float4*)(p1), *(const float4*)(p1 + 4));
}

__device__ __forceinline__ float fast_sigmoid(float x) {
  // single v_exp_f32 + v_rcp_f32 (avoid the IEEE divide expansion)
  return __builtin_amdgcn_rcpf(1.0f + __expf(-x));
}
__device__ __forceinline__ float silu_f(float x) { return x * fast_sigmoid(x); }

// ----------------------------------------------------------------- router ---

__global__ void router_kernel(const float* __restrict__ x, const float* __restrict__ gate_w,
                              const float* __restrict__ e_bias,
                              int* __restrict__ topk_ids, float* __restrict__ topk_w,
                              int* __restrict__ counts) {
  __shared__ float sc[E_EXP];   // sigmoid scores
  __shared__ float sb[E_EXP];   // biased scores
  const int t = blockIdx.x;
  const int e = threadIdx.x;    // 64 threads = 64 experts
  const float* xr = x + (size_t)t * H_DIM;
  const float* wr = gate_w + (size_t)e * H_DIM;
  float acc = 0.f;
  for (int h = 0; h < H_DIM; h += 4) {
    float4 xv = *(const float4*)(xr + h);
    float4 wv = *(const float4*)(wr + h);
    acc += xv.x * wv.x + xv.y * wv.y + xv.z * wv.z + xv.w * wv.w;
  }
  float s = fast_sigmoid(acc);
  sc[e] = s;
  sb[e] = s + e_bias[e];
  __syncthreads();
  if (e == 0) {
    // group scores: sum of top-2 biased scores per group of 8
    float gs[NGROUP];
    for (int g = 0; g < NGROUP; ++g) {
      float m1 = -1e30f, m2 = -1e30f;
      for (int j = 0; j < GSIZE; ++j) {
        float v = sb[g * GSIZE + j];
        if (v > m1) { m2 = m1; m1 = v; } else if (v > m2) { m2 = v; }
      }
      gs[g] = m1 + m2;
    }
    unsigned gmask = 0;
    for (int r = 0; r < TOPKG; ++r) {          // top TOPKG groups
      int best = 0; float bv = -1e30f;
      for (int g = 0; g < NGROUP; ++g)
        if (!((gmask >> g) & 1u) && gs[g] > bv) { bv = gs[g]; best = g; }
      gmask |= 1u << best;
    }
    int ids[TOPK]; float wv[TOPK];
    unsigned long long used = 0ull;
    float wsum = 0.f;
    for (int r = 0; r < TOPK; ++r) {           // top-K experts in allowed groups
      int best = 0; float bv = -1e30f;
      for (int ee = 0; ee < E_EXP; ++ee) {
        if (!((gmask >> (ee / GSIZE)) & 1u)) continue;
        if ((used >> ee) & 1ull) continue;
        if (sb[ee] > bv) { bv = sb[ee]; best = ee; }
      }
      used |= 1ull << best;
      ids[r] = best;
      wv[r]  = sc[best];                       // unbiased score
      wsum  += sc[best];
    }
    float inv = 1.0f / wsum;                   // one divide per token: fine
    for (int r = 0; r < TOPK; ++r) {
      topk_ids[t * TOPK + r] = ids[r];
      topk_w[t * TOPK + r]   = wv[r] * inv;
      atomicAdd(&counts[ids[r]], 1);
    }
  }
}

__global__ void scan_kernel(const int* __restrict__ counts, int* __restrict__ offsets) {
  if (threadIdx.x == 0) {
    int acc = 0;
    for (int e = 0; e < E_EXP; ++e) { offsets[e] = acc; acc += counts[e]; }
    offsets[E_EXP] = acc;
  }
}

// Deterministic CSR build (token order) with wave32 ballot prefix ranks.
__global__ void scatter_kernel(const int* __restrict__ topk_ids, const float* __restrict__ topk_w,
                               const int* __restrict__ offsets,
                               int* __restrict__ pair_token, float* __restrict__ pair_w) {
  const int e = blockIdx.x;
  const int lane = threadIdx.x & 31;
  int pos = offsets[e];
  for (int c = 0; c < T_TOK * TOPK; c += 32) {
    int p = c + lane;
    bool flag = (topk_ids[p] == e);
    unsigned bal = __builtin_amdgcn_ballot_w32(flag);
    if (flag) {
      int rank = __builtin_popcount(bal & ((1u << lane) - 1u));
      pair_token[pos + rank] = p / TOPK;
      pair_w[pos + rank]     = topk_w[p];
    }
    pos += __builtin_popcount(bal);
  }
}

// ------------------------------------------------------- routed expert FFN ---
// Each wave computes a 16x64 strip: one A fragment feeds 8 WMMAs per k-step.
// k-loop unrolled x2 so the scheduler can fill WMMA hazard slots with the
// other iteration's loads/converts instead of v_nop.

__global__ void __launch_bounds__(128)
moe_gateup_kernel(const float* __restrict__ x,
                  const float* __restrict__ w_gate, const float* __restrict__ w_up,
                  const int* __restrict__ offsets, const int* __restrict__ pair_token,
                  float* __restrict__ hbuf) {
  const int e    = blockIdx.x >> 6;          // / (T/16)
  const int tile = blockIdx.x & 63;
  const int base = offsets[e];
  const int ne   = offsets[e + 1] - base;
  const int row0 = tile * 16;
  if (row0 >= ne) return;                    // inactive token tile for this expert
  const int valid = ne - row0;
  const int lane = threadIdx.x & 31;
  const int wave = threadIdx.x >> 5;
  const int nb   = blockIdx.y * 256 + wave * 64;  // 64-wide I strip per wave
  const int hi   = lane >> 4;
  const int m    = lane & 15;

  const int rm  = row0 + ((m < valid) ? m : 0);   // clamp A row (garbage rows unstored)
  const int tok = pair_token[base + rm];
  const float* arow = x + (size_t)tok * H_DIM;
  const float* gr = w_gate + ((size_t)e * I_DIM + nb + m) * H_DIM;  // lane&15 == B's n
  const float* ur = w_up   + ((size_t)e * I_DIM + nb + m) * H_DIM;

  v8f ag[4], au[4];
#pragma unroll
  for (int q = 0; q < 4; ++q) { ag[q] = (v8f){0.f,0.f,0.f,0.f,0.f,0.f,0.f,0.f};
                                au[q] = (v8f){0.f,0.f,0.f,0.f,0.f,0.f,0.f,0.f}; }
#pragma unroll 2
  for (int kb = 0; kb < H_DIM; kb += 32) {
    v16h a = load_a_f32(arow, kb, hi);
    __builtin_prefetch(gr + kb + 64 + hi * 16, 0, 0);   // one unrolled pair ahead
    __builtin_prefetch(ur + kb + 64 + hi * 16, 0, 0);
#pragma unroll
    for (int q = 0; q < 4; ++q) {
      ag[q] = wmma_f16(a, load_b_f32(gr + (size_t)q * 16 * H_DIM, kb, hi), ag[q]);
      au[q] = wmma_f16(a, load_b_f32(ur + (size_t)q * 16 * H_DIM, kb, hi), au[q]);
    }
  }
  const int n0 = nb + (lane & 15);
  if (valid >= 16) {                         // common full-tile fast path
#pragma unroll
    for (int q = 0; q < 4; ++q)
#pragma unroll
      for (int j = 0; j < 8; ++j) {
        int mm = j + hi * 8;                 // C/D layout: M = j + hi*8, N = lane&15
        hbuf[(size_t)(base + row0 + mm) * I_DIM + n0 + q * 16] = silu_f(ag[q][j]) * au[q][j];
      }
  } else {
#pragma unroll
    for (int q = 0; q < 4; ++q)
#pragma unroll
      for (int j = 0; j < 8; ++j) {
        int mm = j + hi * 8;
        if (mm < valid)
          hbuf[(size_t)(base + row0 + mm) * I_DIM + n0 + q * 16] = silu_f(ag[q][j]) * au[q][j];
      }
  }
}

__global__ void __launch_bounds__(128)
moe_down_kernel(const float* __restrict__ hbuf, const float* __restrict__ w_down,
                const int* __restrict__ offsets, const int* __restrict__ pair_token,
                const float* __restrict__ pair_w, float* __restrict__ out) {
  const int e    = blockIdx.x >> 6;
  const int tile = blockIdx.x & 63;
  const int base = offsets[e];
  const int ne   = offsets[e + 1] - base;
  const int row0 = tile * 16;
  if (row0 >= ne) return;
  const int valid = ne - row0;
  const int lane = threadIdx.x & 31;
  const int wave = threadIdx.x >> 5;
  const int nb   = blockIdx.y * 256 + wave * 64;  // 64-wide H strip per wave
  const int hi   = lane >> 4;
  const int m    = lane & 15;

  const int rm = row0 + ((m < valid) ? m : 0);
  const float* arow = hbuf + (size_t)(base + rm) * I_DIM;
  const float* dr   = w_down + ((size_t)e * H_DIM + nb + m) * I_DIM;

  v8f acc[4];
#pragma unroll
  for (int q = 0; q < 4; ++q) acc[q] = (v8f){0.f,0.f,0.f,0.f,0.f,0.f,0.f,0.f};
#pragma unroll 2
  for (int kb = 0; kb < I_DIM; kb += 32) {
    v16h a = load_a_f32(arow, kb, hi);
    __builtin_prefetch(dr + kb + 64 + hi * 16, 0, 0);
#pragma unroll
    for (int q = 0; q < 4; ++q)
      acc[q] = wmma_f16(a, load_b_f32(dr + (size_t)q * 16 * I_DIM, kb, hi), acc[q]);
  }
  const int n0 = nb + (lane & 15);
#pragma unroll
  for (int j = 0; j < 8; ++j) {
    int mm = j + hi * 8;
    if (mm < valid) {
      int pos = base + row0 + mm;
      float w = pair_w[pos] * RSCALE;        // cw * ROUTED_SCALE (linear, commutes)
      float* orow = out + (size_t)pair_token[pos] * H_DIM + n0;
#pragma unroll
      for (int q = 0; q < 4; ++q)
        atomicAdd(orow + q * 16, acc[q][j] * w);
    }
  }
}

// ------------------------------------------------------------ shared expert ---

__global__ void __launch_bounds__(128)
shared_gateup_kernel(const float* __restrict__ x, const float* __restrict__ wgu,
                     float* __restrict__ hs) {
  const int mtile = blockIdx.x;
  const int lane = threadIdx.x & 31;
  const int wave = threadIdx.x >> 5;
  const int nb   = blockIdx.y * 256 + wave * 64;
  const int hi   = lane >> 4;
  const int m    = lane & 15;
  const float* arow = x + (size_t)(mtile * 16 + m) * H_DIM;
  const float* gr = wgu + (size_t)(nb + m) * H_DIM;            // gate rows [0, IS)
  const float* ur = wgu + (size_t)(nb + m + IS_DIM) * H_DIM;   // up rows [IS, 2*IS)

  v8f ag[4], au[4];
#pragma unroll
  for (int q = 0; q < 4; ++q) { ag[q] = (v8f){0.f,0.f,0.f,0.f,0.f,0.f,0.f,0.f};
                                au[q] = (v8f){0.f,0.f,0.f,0.f,0.f,0.f,0.f,0.f}; }
#pragma unroll 2
  for (int kb = 0; kb < H_DIM; kb += 32) {
    v16h a = load_a_f32(arow, kb, hi);
    __builtin_prefetch(gr + kb + 64 + hi * 16, 0, 0);
    __builtin_prefetch(ur + kb + 64 + hi * 16, 0, 0);
#pragma unroll
    for (int q = 0; q < 4; ++q) {
      ag[q] = wmma_f16(a, load_b_f32(gr + (size_t)q * 16 * H_DIM, kb, hi), ag[q]);
      au[q] = wmma_f16(a, load_b_f32(ur + (size_t)q * 16 * H_DIM, kb, hi), au[q]);
    }
  }
  const int n0 = nb + (lane & 15);
#pragma unroll
  for (int q = 0; q < 4; ++q)
#pragma unroll
    for (int j = 0; j < 8; ++j) {
      int t = mtile * 16 + j + hi * 8;
      hs[(size_t)t * IS_DIM + n0 + q * 16] = silu_f(ag[q][j]) * au[q][j];
    }
}

__global__ void __launch_bounds__(128)
shared_down_kernel(const float* __restrict__ hs, const float* __restrict__ wd,
                   float* __restrict__ out) {
  const int mtile = blockIdx.x;
  const int lane = threadIdx.x & 31;
  const int wave = threadIdx.x >> 5;
  const int nb   = blockIdx.y * 256 + wave * 64;
  const int hi   = lane >> 4;
  const int m    = lane & 15;
  const float* arow = hs + (size_t)(mtile * 16 + m) * IS_DIM;
  const float* dr   = wd + (size_t)(nb + m) * IS_DIM;

  v8f acc[4];
#pragma unroll
  for (int q = 0; q < 4; ++q) acc[q] = (v8f){0.f,0.f,0.f,0.f,0.f,0.f,0.f,0.f};
#pragma unroll 2
  for (int kb = 0; kb < IS_DIM; kb += 32) {
    v16h a = load_a_f32(arow, kb, hi);
    __builtin_prefetch(dr + kb + 64 + hi * 16, 0, 0);
#pragma unroll
    for (int q = 0; q < 4; ++q)
      acc[q] = wmma_f16(a, load_b_f32(dr + (size_t)q * 16 * IS_DIM, kb, hi), acc[q]);
  }
  const int n0 = nb + (lane & 15);
#pragma unroll
  for (int q = 0; q < 4; ++q)
#pragma unroll
    for (int j = 0; j < 8; ++j) {
      int t = mtile * 16 + j + hi * 8;
      out[(size_t)t * H_DIM + n0 + q * 16] += acc[q][j];  // unique writer after routed pass
    }
}

// ------------------------------------------------------------------ launch ---

extern "C" void kernel_launch(void* const* d_in, const int* in_sizes, int n_in,
                              void* d_out, int out_size, void* d_ws, size_t ws_size,
                              hipStream_t stream) {
  (void)in_sizes; (void)n_in; (void)ws_size;
  const float* x   = (const float*)d_in[0];   // [T,H]
  const float* gw  = (const float*)d_in[1];   // [E,H]
  const float* eb  = (const float*)d_in[2];   // [E]
  const float* wg  = (const float*)d_in[3];   // [E,I,H]
  const float* wu  = (const float*)d_in[4];   // [E,I,H]
  const float* wd  = (const float*)d_in[5];   // [E,H,I]
  const float* sgu = (const float*)d_in[6];   // [2*IS,H]
  const float* sd  = (const float*)d_in[7];   // [H,IS]
  float* out = (float*)d_out;                 // [T,H] f32

  // workspace layout (~42 MB)
  char* ws = (char*)d_ws;
  int*   counts   = (int*)(ws + 0);                         // 64
  int*   offsets  = (int*)(ws + 1024);                      // 65
  int*   topk_ids = (int*)(ws + 4096);                      // T*K
  float* topk_wv  = (float*)(ws + 4096 + 1 * 32768);        // T*K
  int*   pair_tok = (int*)(ws + 4096 + 2 * 32768);          // T*K (CSR)
  float* pair_wv  = (float*)(ws + 4096 + 3 * 32768);        // T*K (CSR)
  float* hbuf     = (float*)(ws + 4096 + 4 * 32768);        // T*K*I f32 = 32 MB
  float* hs       = (float*)(ws + 4096 + 4 * 32768 +
                             (size_t)T_TOK * TOPK * I_DIM * sizeof(float)); // T*IS = 8 MB

  hipMemsetAsync(counts, 0, E_EXP * sizeof(int), stream);
  hipMemsetAsync(d_out, 0, (size_t)out_size * sizeof(float), stream);

  router_kernel<<<T_TOK, E_EXP, 0, stream>>>(x, gw, eb, topk_ids, topk_wv, counts);
  scan_kernel<<<1, 32, 0, stream>>>(counts, offsets);
  scatter_kernel<<<E_EXP, 32, 0, stream>>>(topk_ids, topk_wv, offsets, pair_tok, pair_wv);

  moe_gateup_kernel<<<dim3(E_EXP * (T_TOK / 16), I_DIM / 256), 128, 0, stream>>>(
      x, wg, wu, offsets, pair_tok, hbuf);
  moe_down_kernel<<<dim3(E_EXP * (T_TOK / 16), H_DIM / 256), 128, 0, stream>>>(
      hbuf, wd, offsets, pair_tok, pair_wv, out);

  shared_gateup_kernel<<<dim3(T_TOK / 16, IS_DIM / 256), 128, 0, stream>>>(x, sgu, hs);
  shared_down_kernel<<<dim3(T_TOK / 16, H_DIM / 256), 128, 0, stream>>>(hs, sd, out);
}
